// Model_2302102471208
// MI455X (gfx1250) — matrix operand order repaired and checked by reading, not compiled, not run on hardware
//
#include <hip/hip_runtime.h>
#include <hip/hip_bf16.h>
#include <math.h>
#include <stdint.h>

// Problem constants (from reference): B=32, S=512, V=32000, D=512, H=256, L=9
#define B_   32
#define S_   512
#define D_   512
#define H_   256
#define L_   9
#define G4H  1024            // 4*H gate width
#define DIN  512             // input width of both LSTM layers (D == 2*H == 512)
#define NTOK (B_ * S_)       // 16384 tokens

typedef __attribute__((ext_vector_type(16))) __bf16 v16bf;
typedef __attribute__((ext_vector_type(8)))  __bf16 v8bf;
typedef __attribute__((ext_vector_type(8)))  float  v8f;
typedef __attribute__((ext_vector_type(4)))  unsigned int v4u;
typedef __attribute__((ext_vector_type(4)))  int v4i;
typedef __attribute__((ext_vector_type(8)))  int v8i;

#define CAT16(lo, hi) __builtin_shufflevector((lo), (hi), 0,1,2,3,4,5,6,7,8,9,10,11,12,13,14,15)

// --- TDM availability / arity detection ------------------------------------
#if defined(__has_builtin)
#if __has_builtin(__builtin_amdgcn_tensor_load_to_lds)
#define HAVE_TDM 1
#endif
#endif
#ifndef HAVE_TDM
#define HAVE_TDM 0
#endif

#if __has_include(<hip/amd_detail/amd_gfx1250_TDM.h>)
#define TDM_SIX_ARG 1       // therock/clang-23 headers -> 6-arg builtin
#else
#define TDM_SIX_ARG 0       // ROCm 7.2 / clang-22 -> 5-arg builtin
#endif

static __device__ __forceinline__ __bf16 to_bf16(float f) {
  unsigned int u = __float_as_uint(f);
  u += 0x7FFFu + ((u >> 16) & 1u);        // round-to-nearest-even
  unsigned short s = (unsigned short)(u >> 16);
  __bf16 r;
  __builtin_memcpy(&r, &s, 2);
  return r;
}

static __device__ __forceinline__ float sigmoidf_(float x) {
  return 1.0f / (1.0f + __expf(-x));
}

static __device__ __forceinline__ void tdm_wait0() {
#if defined(__has_builtin) && __has_builtin(__builtin_amdgcn_s_wait_tensorcnt)
  __builtin_amdgcn_s_wait_tensorcnt(0);
#else
  asm volatile("s_wait_tensorcnt 0x0" ::: "memory");
#endif
}

#if HAVE_TDM
// Issue a 2D TDM load: tile (tile_d0 elements along contiguous dim0) x
// (tile_d1 rows along dim1) of a row-major bf16 tensor into LDS at lds_addr.
// D# packing per CDNA5 ISA ch.8 (groups 2/3 zero => 2D, tile_dim2/3 unused).
static __device__ __forceinline__ void tdm_load_2d_bf16(
    uint32_t lds_addr, const void* gptr,
    uint32_t tensor_d0, uint32_t tensor_d1,
    uint32_t tile_d0, uint32_t tile_d1,
    uint64_t stride0) {
  uint64_t ga = (uint64_t)(uintptr_t)gptr;
  uint64_t stride1 = stride0;                       // unused for 2D tiles
  v4u g0;
  g0[0] = 1u;                                       // count=1, user descriptor
  g0[1] = lds_addr;                                 // LDS byte address
  g0[2] = (uint32_t)ga;                             // global_addr[31:0]
  g0[3] = (uint32_t)((ga >> 32) & 0x01FFFFFFu)      // global_addr[56:32]
        | 0x80000000u;                              // type=2 ("image")
  v8i g1;
  g1[0] = (int)(1u << 16);                          // data_size=1 -> 2 bytes
  g1[1] = (int)((tensor_d0 & 0xFFFFu) << 16);       // tensor_dim0[15:0]
  g1[2] = (int)(((tensor_d0 >> 16) & 0xFFFFu)       // tensor_dim0[31:16]
        |        ((tensor_d1 & 0xFFFFu) << 16));    // tensor_dim1[15:0]
  g1[3] = (int)(((tensor_d1 >> 16) & 0xFFFFu)       // tensor_dim1[31:16]
        |        ((tile_d0 & 0xFFFFu) << 16));      // tile_dim0
  g1[4] = (int)(tile_d1 & 0xFFFFu);                 // tile_dim1 (tile_dim2=0)
  g1[5] = (int)(uint32_t)(stride0 & 0xFFFFFFFFu);   // dim0_stride[31:0]
  g1[6] = (int)(uint32_t)(((stride0 >> 32) & 0xFFFFu)
        |        ((stride1 & 0xFFFFu) << 16));      // dim0_str[47:32]|dim1_str[15:0]
  g1[7] = (int)(uint32_t)((stride1 >> 16) & 0xFFFFFFFFu); // dim1_stride[47:16]
  v4i z4 = {0, 0, 0, 0};
#if TDM_SIX_ARG
  v8i z8 = {0, 0, 0, 0, 0, 0, 0, 0};
  __builtin_amdgcn_tensor_load_to_lds(g0, g1, z4, z4, z8, 0);
#else
  __builtin_amdgcn_tensor_load_to_lds(g0, g1, z4, z4, 0);
#endif
}
#endif // HAVE_TDM

// ---------------------------------------------------------------------------
// f32 -> bf16 weight conversion
// ---------------------------------------------------------------------------
__global__ void f32_to_bf16_kernel(const float* __restrict__ src,
                                   __bf16* __restrict__ dst, int n) {
  int i = blockIdx.x * blockDim.x + threadIdx.x;
  if (i < n) dst[i] = to_bf16(src[i]);
}

// ---------------------------------------------------------------------------
// Embedding gather (masked), stored bf16, row index = s*B + b
// ---------------------------------------------------------------------------
__global__ void embed_kernel(const int* __restrict__ ids,
                             const int* __restrict__ mask,
                             const float* __restrict__ emb,
                             __bf16* __restrict__ xbf) {
  int tok = blockIdx.x;                  // b*S + s
  int b = tok / S_, s = tok % S_;
  int id = ids[tok];
  float m = (float)mask[tok];
  size_t row = (size_t)(s * B_ + b) * DIN;
  const float* e = emb + (size_t)id * D_;
  for (int d = threadIdx.x; d < D_; d += blockDim.x)
    xbf[row + d] = to_bf16(e[d] * m);
}

// ---------------------------------------------------------------------------
// Batched input projection GEMM with TDM-staged LDS panels:
//   G[m][n] = X[m][:] . W[n][:] + bih[n] + bhh[n],  M=16384, N=1024, K=512.
// Block (256 thr = 8 waves) computes a 64x32 output tile.  Per K-chunk (256):
// wave 0 issues two tensor_load_to_lds DMAs (A panel 64x256 bf16 = 32KB,
// B panel 32x256 bf16 = 16KB), s_wait_tensorcnt, barrier, then each wave runs
// 8 x v_wmma_f32_16x16x32_bf16 with fragments from LDS (ds_load_b128).
// ---------------------------------------------------------------------------
#define KCHUNK 256
#define A_ELEMS (64 * KCHUNK)            // 16384 bf16 = 32KB
#define B_ELEMS (32 * KCHUNK)            //  8192 bf16 = 16KB

__global__ void __launch_bounds__(256) igemm_kernel(
    const __bf16* __restrict__ X,
    const __bf16* __restrict__ W,
    const float* __restrict__ bih,
    const float* __restrict__ bhh,
    float* __restrict__ G) {
  __shared__ __align__(16) __bf16 lds[A_ELEMS + B_ELEMS];   // 48 KB

  const int wave = threadIdx.x >> 5;
  const int lane = threadIdx.x & 31;
  const int half = lane >> 4;
  const int lm   = lane & 15;
  const int mblk = blockIdx.x >> 5;      // 0..255  -> rows base mblk*64
  const int nblk = blockIdx.x & 31;      // 0..31   -> cols base nblk*32
  const int mt   = wave >> 1;            // 0..3 local mtile
  const int nt   = wave & 1;             // 0..1 local ntile
  const int row0 = mblk * 64;
  const int col0 = nblk * 32;

  const __bf16* apan = &lds[0];
  const __bf16* bpan = &lds[A_ELEMS];

  v8f acc = {};
  for (int kc = 0; kc < DIN; kc += KCHUNK) {
#if HAVE_TDM
    if (wave == 0) {
      uint32_t lds0 = (uint32_t)(uintptr_t)&lds[0];
      tdm_load_2d_bf16(lds0, X + (size_t)row0 * DIN + kc,
                       /*tensor_d0=*/DIN, /*tensor_d1=*/NTOK,
                       /*tile_d0=*/KCHUNK, /*tile_d1=*/64, /*stride0=*/DIN);
      tdm_load_2d_bf16(lds0 + A_ELEMS * 2, W + (size_t)col0 * DIN + kc,
                       /*tensor_d0=*/DIN, /*tensor_d1=*/G4H,
                       /*tile_d0=*/KCHUNK, /*tile_d1=*/32, /*stride0=*/DIN);
      tdm_wait0();
    }
    __syncthreads();                     // panels visible to all waves
#else
    for (int i = threadIdx.x * 8; i < A_ELEMS; i += 256 * 8) {
      int r = i / KCHUNK, k = i % KCHUNK;
      *(v8bf*)(&lds[r * KCHUNK + k]) =
          *(const v8bf*)(X + (size_t)(row0 + r) * DIN + kc + k);
    }
    for (int i = threadIdx.x * 8; i < B_ELEMS; i += 256 * 8) {
      int r = i / KCHUNK, k = i % KCHUNK;
      *(v8bf*)(&lds[A_ELEMS + r * KCHUNK + k]) =
          *(const v8bf*)(W + (size_t)(col0 + r) * DIN + kc + k);
    }
    __syncthreads();
#endif

    const int ar = mt * 16 + lm;         // A panel row for this lane
    const int br = nt * 16 + lm;         // B panel row (= output column)
#pragma unroll
    for (int k0 = 0; k0 < KCHUNK; k0 += 32) {
      v8bf a0 = *(const v8bf*)(apan + ar * KCHUNK + k0 + half * 8);
      v8bf a1 = *(const v8bf*)(apan + ar * KCHUNK + k0 + 16 + half * 8);
      v16bf Af = CAT16(a0, a1);
      v8bf b0 = *(const v8bf*)(bpan + br * KCHUNK + k0 + half * 8);
      v8bf b1 = *(const v8bf*)(bpan + br * KCHUNK + k0 + 16 + half * 8);
      v16bf Bf = CAT16(b0, b1);
      acc = __builtin_amdgcn_wmma_f32_16x16x32_bf16(false, Af, false, Bf,
                                                    (short)0, acc, false, false);
    }
    __syncthreads();                     // done reading before next DMA chunk
  }

  int n = col0 + nt * 16 + lm;
  float bias = bih[n] + bhh[n];
  size_t base = (size_t)(row0 + mt * 16 + half * 8) * G4H + n;
#pragma unroll
  for (int v = 0; v < 8; ++v)
    G[base + (size_t)v * G4H] = acc[v] + bias;
}

// ---------------------------------------------------------------------------
// Sequential LSTM over S steps. grid = 2 (dir), block = 1024 (32 waves).
// Wave w: mtile = w/16 (batch rows), g16 = w%16 (16 hidden cols). Each wave
// computes the i,f,g,o gate tiles for its (16 batch x 16 hidden) block so the
// cell state c stays in 8 accumulator VGPRs for the whole sequence.
// h_{t-1} lives in LDS as bf16 (A fragments via ds_load_b128).
// ---------------------------------------------------------------------------
__global__ void __launch_bounds__(1024) lstm_kernel(
    const float* __restrict__ G,        // [2][NTOK][1024] gate pre-activations
    const __bf16* __restrict__ Whh,     // [2][1024][256] bf16
    __bf16* __restrict__ out_bf,        // [NTOK][512] (dir cols) or nullptr
    float* __restrict__ out_f)          // [NTOK][512] or nullptr
{
  __shared__ __align__(16) __bf16 hbuf[B_ * H_];   // 16 KB

  const int dir   = blockIdx.x;
  const int wave  = threadIdx.x >> 5;
  const int mtile = wave >> 4;                     // 0/1
  const int g16   = wave & 15;                     // hidden column block
  const int lane  = threadIdx.x & 31;
  const int half  = lane >> 4;
  const int lm    = lane & 15;

  for (int i = threadIdx.x; i < B_ * H_; i += blockDim.x) hbuf[i] = to_bf16(0.f);
  __syncthreads();

  const float*  Gd = G   + (size_t)dir * NTOK * G4H;
  const __bf16* Wd = Whh + (size_t)dir * G4H * H_;

  float c[8];
#pragma unroll
  for (int v = 0; v < 8; ++v) c[v] = 0.f;

  const int hcol = g16 * 16 + lm;                  // hidden index 0..255
  const int arow = mtile * 16 + lm;                // batch row for A fragment

  for (int t = 0; t < S_; ++t) {
    int s = dir ? (S_ - 1 - t) : t;
    size_t grow0 = ((size_t)s * B_ + mtile * 16 + half * 8) * G4H;

    if (t + 1 < S_) {                              // prefetch next step's G row
      int sn = dir ? (S_ - 2 - t) : (t + 1);
      __builtin_prefetch(Gd + ((size_t)sn * B_ + mtile * 16 + half * 8) * G4H + hcol, 0, 3);
    }

    v8f acc[4];
#pragma unroll
    for (int gi = 0; gi < 4; ++gi) {
      int n = gi * H_ + hcol;                      // gate column in [0,1024)
      v8f a;
#pragma unroll
      for (int v = 0; v < 8; ++v)
        a[v] = Gd[grow0 + (size_t)v * G4H + n];    // C init = gate pre-activation
      const __bf16* wr = Wd + (size_t)n * H_;
#pragma unroll
      for (int kk = 0; kk < 8; ++kk) {             // K = 256 in 8 x 32 steps
        int kb = kk * 32;
        v8bf a0 = *(const v8bf*)(&hbuf[arow * H_ + kb + half * 8]);
        v8bf a1 = *(const v8bf*)(&hbuf[arow * H_ + kb + 16 + half * 8]);
        v16bf Af = CAT16(a0, a1);
        v8bf b0 = *(const v8bf*)(wr + kb + half * 8);
        v8bf b1 = *(const v8bf*)(wr + kb + 16 + half * 8);
        v16bf Bf = CAT16(b0, b1);
        a = __builtin_amdgcn_wmma_f32_16x16x32_bf16(false, Af, false, Bf,
                                                    (short)0, a, false, false);
      }
      acc[gi] = a;
    }
    __syncthreads();                               // all hbuf reads done

#pragma unroll
    for (int v = 0; v < 8; ++v) {
      float iv = sigmoidf_(acc[0][v]);
      float fv = sigmoidf_(acc[1][v]);
      float gv = tanhf(acc[2][v]);
      float ov = sigmoidf_(acc[3][v]);
      c[v] = fv * c[v] + iv * gv;
      float hv = ov * tanhf(c[v]);
      int b = mtile * 16 + half * 8 + v;
      hbuf[b * H_ + hcol] = to_bf16(hv);
      size_t orow = ((size_t)s * B_ + b) * (2 * H_) + (size_t)dir * H_ + hcol;
      if (out_bf) out_bf[orow] = to_bf16(hv);
      if (out_f)  out_f[orow]  = hv;
    }
    __syncthreads();                               // new h visible to all waves
  }
}

// ---------------------------------------------------------------------------
// LayerNorm over 512 dims + 9-way classifier.  One block per token.
// ---------------------------------------------------------------------------
__global__ void ln_cls_kernel(const float* __restrict__ Y,
                              const float* __restrict__ gamma,
                              const float* __restrict__ beta,
                              const float* __restrict__ cw,
                              const float* __restrict__ cb,
                              float* __restrict__ logits) {
  __shared__ float norm[2 * H_];
  __shared__ float red[256];
  int tok = blockIdx.x;                  // b*S + s
  int b = tok / S_, s = tok % S_;
  const float* row = Y + (size_t)(s * B_ + b) * (2 * H_);
  int tid = threadIdx.x;
  float x0 = row[tid], x1 = row[tid + 256];
  red[tid] = x0 + x1;
  __syncthreads();
  for (int st = 128; st > 0; st >>= 1) {
    if (tid < st) red[tid] += red[tid + st];
    __syncthreads();
  }
  float mu = red[0] * (1.0f / 512.0f);
  __syncthreads();
  float d0 = x0 - mu, d1 = x1 - mu;
  red[tid] = d0 * d0 + d1 * d1;
  __syncthreads();
  for (int st = 128; st > 0; st >>= 1) {
    if (tid < st) red[tid] += red[tid + st];
    __syncthreads();
  }
  float inv = rsqrtf(red[0] * (1.0f / 512.0f) + 1e-5f);
  norm[tid]       = d0 * inv * gamma[tid]       + beta[tid];
  norm[tid + 256] = d1 * inv * gamma[tid + 256] + beta[tid + 256];
  __syncthreads();
  if (tid < L_) {
    float acc = cb[tid];
    const float* w = cw + (size_t)tid * (2 * H_);
    for (int d = 0; d < 2 * H_; ++d) acc += norm[d] * w[d];
    logits[(size_t)tok * L_ + tid] = acc;
  }
}

// ---------------------------------------------------------------------------
// CRF NLL: forward algorithm (logZ) + gold score, loss = mean(logZ - gold).
// One block of 288 threads = (b, j) pairs, 32 batches x 9 tags.
// ---------------------------------------------------------------------------
__global__ void crf_kernel(const float* __restrict__ logits,
                           const int* __restrict__ labels,
                           const int* __restrict__ lens,
                           const float* __restrict__ trans,   // (11,11)
                           float* __restrict__ out_loss) {
  __shared__ float alpha[B_ * L_];
  __shared__ float nxt[B_ * L_];
  __shared__ float logZ[B_];
  __shared__ float gold[B_];
  const int T11 = L_ + 2;                // 11
  int tid = threadIdx.x;
  int b = tid / L_, j = tid % L_;
  bool act = tid < B_ * L_;

  if (act)
    alpha[tid] = trans[L_ * T11 + j] + logits[((size_t)b * S_) * L_ + j]; // START row
  __syncthreads();

  for (int t = 1; t < S_; ++t) {
    if (act) {
      float mx = -1e30f;
#pragma unroll
      for (int i = 0; i < L_; ++i)
        mx = fmaxf(mx, alpha[b * L_ + i] + trans[i * T11 + j]);
      float sum = 0.f;
#pragma unroll
      for (int i = 0; i < L_; ++i)
        sum += __expf(alpha[b * L_ + i] + trans[i * T11 + j] - mx);
      float newv = mx + __logf(sum) + logits[((size_t)b * S_ + t) * L_ + j];
      nxt[tid] = (t < lens[b]) ? newv : alpha[tid];
    }
    __syncthreads();
    if (act) alpha[tid] = nxt[tid];
    __syncthreads();
  }

  if (act && j == 0) {
    float mx = -1e30f;
    for (int i = 0; i < L_; ++i)
      mx = fmaxf(mx, alpha[b * L_ + i] + trans[i * T11 + (L_ + 1)]);   // STOP col
    float sum = 0.f;
    for (int i = 0; i < L_; ++i)
      sum += __expf(alpha[b * L_ + i] + trans[i * T11 + (L_ + 1)] - mx);
    logZ[b] = mx + __logf(sum);

    int len = lens[b];
    const int* lab = labels + (size_t)b * S_;
    int prev = lab[0];
    float g = trans[L_ * T11 + prev] + logits[((size_t)b * S_) * L_ + prev];
    for (int t = 1; t < S_; ++t) {
      if (t < len) {
        int cur = lab[t];
        g += trans[prev * T11 + cur] + logits[((size_t)b * S_ + t) * L_ + cur];
        prev = cur;
      }
    }
    g += trans[prev * T11 + (L_ + 1)];
    gold[b] = g;
  }
  __syncthreads();
  if (tid == 0) {
    float acc = 0.f;
    for (int bb = 0; bb < B_; ++bb) acc += logZ[bb] - gold[bb];
    out_loss[0] = acc / (float)B_;
  }
}

// ---------------------------------------------------------------------------
// Host launcher
// ---------------------------------------------------------------------------
extern "C" void kernel_launch(void* const* d_in, const int* in_sizes, int n_in,
                              void* d_out, int out_size, void* d_ws, size_t ws_size,
                              hipStream_t stream) {
  const int*   ids    = (const int*)d_in[0];
  const int*   amask  = (const int*)d_in[1];
  const int*   lens   = (const int*)d_in[2];
  const int*   labels = (const int*)d_in[3];
  const float* emb    = (const float*)d_in[4];
  const float* w_ih   = (const float*)d_in[5];   // (2,2,1024,512)
  const float* w_hh   = (const float*)d_in[6];   // (2,2,1024,256)
  const float* b_ih   = (const float*)d_in[7];   // (2,2,1024)
  const float* b_hh   = (const float*)d_in[8];
  const float* gamma  = (const float*)d_in[9];
  const float* beta   = (const float*)d_in[10];
  const float* cls_w  = (const float*)d_in[11];  // (9,512)
  const float* cls_b  = (const float*)d_in[12];
  const float* trans  = (const float*)d_in[13];  // (11,11)

  // Workspace partition
  char* ws = (char*)d_ws;
  size_t off = 0;
  auto take = [&](size_t bytes) -> char* {
    char* p = ws + off;
    off = (off + bytes + 255) & ~(size_t)255;
    return p;
  };
  __bf16* wih_bf = (__bf16*)take((size_t)2 * 2 * G4H * DIN * 2);  //   4 MB
  __bf16* whh_bf = (__bf16*)take((size_t)2 * 2 * G4H * H_ * 2);   //   2 MB
  __bf16* x_bf   = (__bf16*)take((size_t)NTOK * DIN * 2);         //  16 MB
  __bf16* y1_bf  = (__bf16*)take((size_t)NTOK * DIN * 2);         //  16 MB
  float*  y2_f   = (float*) take((size_t)NTOK * DIN * 4);         //  32 MB
  float*  Gbuf   = (float*) take((size_t)2 * NTOK * G4H * 4);     // 128 MB

  // Convert weights to bf16
  {
    int n1 = 2 * 2 * G4H * DIN;
    int n2 = 2 * 2 * G4H * H_;
    f32_to_bf16_kernel<<<(n1 + 255) / 256, 256, 0, stream>>>(w_ih, wih_bf, n1);
    f32_to_bf16_kernel<<<(n2 + 255) / 256, 256, 0, stream>>>(w_hh, whh_bf, n2);
  }

  // Masked embedding gather -> bf16, row = s*B + b
  embed_kernel<<<NTOK, 256, 0, stream>>>(ids, amask, emb, x_bf);

  for (int layer = 0; layer < 2; ++layer) {
    const __bf16* Xin = layer ? y1_bf : x_bf;
    for (int dir = 0; dir < 2; ++dir) {
      int ld = layer * 2 + dir;
      // 256 row-blocks x 32 col-blocks = 8192 blocks, each 64x32 of G
      igemm_kernel<<<8192, 256, 0, stream>>>(
          Xin, wih_bf + (size_t)ld * G4H * DIN,
          b_ih + (size_t)ld * G4H, b_hh + (size_t)ld * G4H,
          Gbuf + (size_t)dir * NTOK * G4H);
    }
    lstm_kernel<<<2, 1024, 0, stream>>>(
        Gbuf, whh_bf + (size_t)layer * 2 * G4H * H_,
        layer == 0 ? y1_bf : (__bf16*)nullptr,
        layer == 1 ? y2_f : (float*)nullptr);
  }

  float* out = (float*)d_out;
  float* logits = out + 1;                          // out[0] = loss
  ln_cls_kernel<<<NTOK, 256, 0, stream>>>(y2_f, gamma, beta, cls_w, cls_b, logits);
  crf_kernel<<<1, 288, 0, stream>>>(logits, labels, lens, trans, out);

  (void)in_sizes; (void)n_in; (void)out_size; (void)ws_size;
}
